// MultiSelfAttentionHeadVarlenFlex_53609781789410
// MI455X (gfx1250) — compile-verified, matrix operand order checked by reading, not run on hardware
//
#include <hip/hip_runtime.h>
#include <hip/hip_bf16.h>
#include <stdint.h>

// ---------------------------------------------------------------------------
// CDNA5 (gfx1250) bf16 WMMA pipeline for the masked flex-attention block.
// Mask (q//16 == kv%16) is only satisfiable for q < 256, so attention and the
// output projection only touch 256 query rows; rows >= 256 are pure residual.
// Async global->LDS copies (ASYNCcnt) + LDS double buffering in the GEMM.
// ---------------------------------------------------------------------------

typedef __attribute__((ext_vector_type(16))) __bf16 v16bf;
typedef __attribute__((ext_vector_type(8)))  float  v8f;
typedef __attribute__((ext_vector_type(4)))  int    v4i;

#define GEMM_STRIDE 40   // halfs per LDS row: 32 data + 8 pad (80B -> conflict-free b128)
#define ATT_STRIDE 136   // halfs per LDS row for 128-wide tiles (272B, 16B-aligned)

#if __has_builtin(__builtin_amdgcn_global_load_async_to_lds_b128)
#define HAS_ASYNC 1
#else
#define HAS_ASYNC 0
#endif

template <int N>
__device__ __forceinline__ void wait_async() {
#if __has_builtin(__builtin_amdgcn_s_wait_asynccnt)
  __builtin_amdgcn_s_wait_asynccnt(N);
#else
  asm volatile("s_wait_asynccnt %0" :: "i"(N) : "memory");
#endif
}

// 16-byte global -> LDS copy: async (direct-to-LDS, ASYNCcnt) when available.
// Builtin params (from toolchain diagnostics): (AS1 int4* src, AS3 int4* dst,
// imm offset, imm cpol).
__device__ __forceinline__ void copy16_g2l(const __bf16* g, __bf16* l) {
#if HAS_ASYNC
  __builtin_amdgcn_global_load_async_to_lds_b128(
      (__attribute__((address_space(1))) v4i*)(g),
      (__attribute__((address_space(3))) v4i*)(l), 0, 0);
#else
  *(uint4*)l = *(const uint4*)g;
#endif
}

__device__ __forceinline__ v8f wmma_bf16(v16bf a, v16bf b, v8f c) {
  // D = A(16x32) * B(32x16) + C, fp32 accumulate
  return __builtin_amdgcn_wmma_f32_16x16x32_bf16(
      /*neg_a=*/false, a, /*neg_b=*/false, b,
      /*c_mod=*/(short)0, c, /*reuse_a=*/false, /*reuse_b=*/false);
}

union FragU { uint4 q[2]; v16bf v; };

// Fragment loader for A-style operands (and B given as B^T, row = n):
// lane L (row = L&15, half = L>>4) reads K-chunks [half*8, half*8+7] and
// [16+half*8, 16+half*8+7] as two b128 LDS loads.
__device__ __forceinline__ v16bf load_frag(const __bf16* base, int lane, int strideHalves) {
  const int row  = lane & 15;
  const int half = lane >> 4;
  const __bf16* p = base + row * strideHalves + half * 8;
  FragU u;
  u.q[0] = *(const uint4*)(p);
  u.q[1] = *(const uint4*)(p + 16);
  return u.v;
}

// ---------------------------------------------------------------------------
// Weight fp32 -> bf16 with transpose: Wt[n][k] = (bf16)W[k][n]
// ---------------------------------------------------------------------------
__global__ __launch_bounds__(256) void transpose_to_bf16(
    const float* __restrict__ W, __bf16* __restrict__ Wt, int D) {
  __shared__ float tile[32][33];
  const int bx = blockIdx.x * 32;   // column block of W (-> row block of Wt)
  const int by = blockIdx.y * 32;   // row block of W
  const int tx = threadIdx.x;       // 0..31
  const int ty = threadIdx.y;       // 0..7
#pragma unroll
  for (int j = ty; j < 32; j += 8)
    tile[j][tx] = W[(size_t)(by + j) * D + (bx + tx)];
  __syncthreads();
#pragma unroll
  for (int j = ty; j < 32; j += 8)
    Wt[(size_t)(bx + j) * D + (by + tx)] = (__bf16)tile[tx][j];
}

// ---------------------------------------------------------------------------
// Row LayerNorm of x (2048 cols) -> bf16
// ---------------------------------------------------------------------------
__global__ __launch_bounds__(256) void ln_rows_bf16(
    const float* __restrict__ X, __bf16* __restrict__ Y, int D) {
  const int row = blockIdx.x;
  const int tid = threadIdx.x;
  const float* xr = X + (size_t)row * D;
  float s = 0.f, s2 = 0.f;
  for (int c = tid; c < D; c += 256) {
    float v = xr[c];
    s += v; s2 += v * v;
  }
  __shared__ float rs[256], rq[256];
  rs[tid] = s; rq[tid] = s2;
  __syncthreads();
  for (int o = 128; o > 0; o >>= 1) {
    if (tid < o) { rs[tid] += rs[tid + o]; rq[tid] += rq[tid + o]; }
    __syncthreads();
  }
  const float mean = rs[0] / (float)D;
  const float var  = rq[0] / (float)D - mean * mean;
  const float rstd = rsqrtf(var + 1e-5f);
  for (int c = tid; c < D; c += 256)
    Y[(size_t)row * D + c] = (__bf16)((xr[c] - mean) * rstd);
}

// ---------------------------------------------------------------------------
// Per-head (DH=128) LayerNorm -> bf16. One wave per (row, head).
// ---------------------------------------------------------------------------
__global__ __launch_bounds__(256) void head_ln_bf16(
    const float* __restrict__ X, __bf16* __restrict__ Y, int rows) {
  const int w    = (blockIdx.x * 256 + threadIdx.x) >> 5;
  const int lane = threadIdx.x & 31;
  const int row  = w >> 4;
  const int head = w & 15;
  if (row >= rows) return;
  const float* p = X + (size_t)row * 2048 + head * 128;
  float4 v = *(const float4*)(p + lane * 4);
  float s  = v.x + v.y + v.z + v.w;
  float s2 = v.x * v.x + v.y * v.y + v.z * v.z + v.w * v.w;
#pragma unroll
  for (int m = 16; m > 0; m >>= 1) {
    s  += __shfl_xor(s, m);
    s2 += __shfl_xor(s2, m);
  }
  const float mean = s  * (1.0f / 128.0f);
  const float var  = s2 * (1.0f / 128.0f) - mean * mean;
  const float rstd = rsqrtf(var + 1e-5f);
  union { __bf16 h[4]; uint2 u; } o;
  o.h[0] = (__bf16)((v.x - mean) * rstd);
  o.h[1] = (__bf16)((v.y - mean) * rstd);
  o.h[2] = (__bf16)((v.z - mean) * rstd);
  o.h[3] = (__bf16)((v.w - mean) * rstd);
  *(uint2*)(Y + (size_t)row * 2048 + head * 128 + lane * 4) = o.u;
}

// ---------------------------------------------------------------------------
// WMMA GEMM: C[M x N] (+)= A[M x K] * B, with B given transposed (Bt[N x K]).
// 128x128 block tile, BK=32, 8 waves (4 M x 2 N), each wave 32x64.
// Double-buffered LDS; async global->LDS (ASYNCcnt) when the builtin exists.
// ---------------------------------------------------------------------------
__device__ __forceinline__ void stage_pair(
    const __bf16* __restrict__ A, const __bf16* __restrict__ Bt,
    __bf16* As, __bf16* Bs, int blockM, int blockN, int K, int k0, int tid) {
  // 128 rows x 32 halfs per tile = 512 uint4; each thread moves slots tid, tid+256.
#pragma unroll
  for (int s = 0; s < 2; ++s) {
    const int i = tid + s * 256;
    const int r = i >> 2;
    const int c = (i & 3) << 3;
    copy16_g2l(&A [(size_t)(blockM + r) * K + k0 + c], &As[r * GEMM_STRIDE + c]);
    copy16_g2l(&Bt[(size_t)(blockN + r) * K + k0 + c], &Bs[r * GEMM_STRIDE + c]);
  }
}

__global__ __launch_bounds__(256) void gemm_bf16_nt(
    const __bf16* __restrict__ A, const __bf16* __restrict__ Bt,
    float* __restrict__ C, int M, int N, int K, int addTo) {
  __shared__ __bf16 As[2][128 * GEMM_STRIDE];
  __shared__ __bf16 Bs[2][128 * GEMM_STRIDE];
  const int tid  = threadIdx.x;
  const int wid  = tid >> 5;
  const int lane = tid & 31;
  const int m0 = (wid >> 1) * 32;
  const int n0 = (wid & 1) * 64;
  const int blockM = blockIdx.y * 128;
  const int blockN = blockIdx.x * 128;

  v8f acc[2][4];
#pragma unroll
  for (int i = 0; i < 2; ++i)
#pragma unroll
    for (int j = 0; j < 4; ++j)
      acc[i][j] = (v8f){0.f, 0.f, 0.f, 0.f, 0.f, 0.f, 0.f, 0.f};

  const int nk = K >> 5;
  stage_pair(A, Bt, As[0], Bs[0], blockM, blockN, K, 0, tid);

  for (int kt = 0; kt < nk; ++kt) {
    const int cur = kt & 1;
    const bool more = (kt + 1) < nk;
    if (more)
      stage_pair(A, Bt, As[cur ^ 1], Bs[cur ^ 1], blockM, blockN, K, (kt + 1) << 5, tid);
#if HAS_ASYNC
    // Async loads complete in order: <=4 outstanding means the current tile's
    // 4 per-wave async instructions have landed in LDS.
    if (more) wait_async<4>(); else wait_async<0>();
#endif
    __syncthreads();

    v16bf af[2], bf[4];
#pragma unroll
    for (int i = 0; i < 2; ++i)
      af[i] = load_frag(&As[cur][(m0 + i * 16) * GEMM_STRIDE], lane, GEMM_STRIDE);
#pragma unroll
    for (int j = 0; j < 4; ++j)
      bf[j] = load_frag(&Bs[cur][(n0 + j * 16) * GEMM_STRIDE], lane, GEMM_STRIDE);
#pragma unroll
    for (int i = 0; i < 2; ++i)
#pragma unroll
      for (int j = 0; j < 4; ++j)
        acc[i][j] = wmma_bf16(af[i], bf[j], acc[i][j]);

    __syncthreads();  // all reads of As/Bs[cur] done before it is restaged
  }

  const int sub     = lane & 15;
  const int halfsel = lane >> 4;
#pragma unroll
  for (int i = 0; i < 2; ++i)
#pragma unroll
    for (int j = 0; j < 4; ++j)
#pragma unroll
      for (int r = 0; r < 8; ++r) {
        const int row = blockM + m0 + i * 16 + r + 8 * halfsel;
        const int col = blockN + n0 + j * 16 + sub;
        const size_t idx = (size_t)row * N + col;
        if (addTo) C[idx] += acc[i][j][r];
        else       C[idx]  = acc[i][j][r];
      }
}

// ---------------------------------------------------------------------------
// Attention for one (query block b, head h): 16 queries x 128 keys (kv = 16j+b)
// scores = Q*K^T/sqrt(128) -> softmax -> O = P*V. All MMs via bf16 WMMA.
// Q/K tiles staged with async global->LDS copies when available.
// ---------------------------------------------------------------------------
__global__ __launch_bounds__(256) void attention_kernel(
    const __bf16* __restrict__ Qb,   // 256 x 2048 (head-normed, bf16)
    const __bf16* __restrict__ Kb,   // 2048 x 2048 (head-normed, bf16)
    const float*  __restrict__ Vf,   // 2048 x 2048 fp32
    __bf16* __restrict__ Ob) {       // 256 x 2048 bf16
  const int b = blockIdx.x;          // query block (q rows 16b..16b+15)
  const int h = blockIdx.y;          // head

  __shared__ __bf16 Qs[16  * ATT_STRIDE];
  __shared__ __bf16 Ks[128 * ATT_STRIDE];   // rows = key, cols = dh
  __shared__ __bf16 Vt[128 * ATT_STRIDE];   // rows = dh,  cols = key (V^T)
  __shared__ float  S[16 * 128];
  __shared__ __bf16 P[16 * ATT_STRIDE];

  const int tid  = threadIdx.x;
  const int wid  = tid >> 5;
  const int lane = tid & 31;

  // Q tile: 16 rows x 128 halfs = 256 uint4, one per thread
  {
    const int r = tid >> 4;
    const int c = (tid & 15) * 8;
    copy16_g2l(&Qb[(size_t)(b * 16 + r) * 2048 + h * 128 + c], &Qs[r * ATT_STRIDE + c]);
  }
  // K tile: key j lives at global row 16j+b; 2048 uint4, 8 per thread
  for (int i = tid; i < 2048; i += 256) {
    const int r = i >> 4;
    const int c = (i & 15) * 8;
    copy16_g2l(&Kb[(size_t)(16 * r + b) * 2048 + h * 128 + c], &Ks[r * ATT_STRIDE + c]);
  }
  // V tile, transposed into Vt[dh][key] while converting fp32 -> bf16
  for (int i = tid; i < 4096; i += 256) {
    const int j = i >> 5;          // key index
    const int d = (i & 31) * 4;    // dh base
    float4 v = *(const float4*)&Vf[(size_t)(16 * j + b) * 2048 + h * 128 + d];
    Vt[(d + 0) * ATT_STRIDE + j] = (__bf16)v.x;
    Vt[(d + 1) * ATT_STRIDE + j] = (__bf16)v.y;
    Vt[(d + 2) * ATT_STRIDE + j] = (__bf16)v.z;
    Vt[(d + 3) * ATT_STRIDE + j] = (__bf16)v.w;
  }
#if HAS_ASYNC
  wait_async<0>();
#endif
  __syncthreads();

  // Scores: wave wid owns key tile [wid*16, wid*16+16); K loop over dh (4x32)
  v8f acc = (v8f){0.f, 0.f, 0.f, 0.f, 0.f, 0.f, 0.f, 0.f};
#pragma unroll
  for (int kc = 0; kc < 4; ++kc) {
    v16bf a  = load_frag(&Qs[kc * 32], lane, ATT_STRIDE);
    v16bf bb = load_frag(&Ks[(wid * 16) * ATT_STRIDE + kc * 32], lane, ATT_STRIDE);
    acc = wmma_bf16(a, bb, acc);
  }
  const float scale = 0.08838834764831845f;  // 1/sqrt(128)
  const int sub  = lane & 15;
  const int half = lane >> 4;
#pragma unroll
  for (int r = 0; r < 8; ++r)
    S[(r + 8 * half) * 128 + wid * 16 + sub] = acc[r] * scale;
  __syncthreads();

  // Softmax over 128 keys. Wave wid handles rows 2*wid + half; 16 lanes/row,
  // 8 keys per lane; xor-shuffle masks <16 stay within each half-wave.
  {
    const int row = 2 * wid + half;
    float mx = -1e30f;
#pragma unroll
    for (int t = 0; t < 8; ++t) mx = fmaxf(mx, S[row * 128 + sub * 8 + t]);
#pragma unroll
    for (int m = 8; m > 0; m >>= 1) mx = fmaxf(mx, __shfl_xor(mx, m));
    float e[8];
    float sum = 0.f;
#pragma unroll
    for (int t = 0; t < 8; ++t) {
      e[t] = __expf(S[row * 128 + sub * 8 + t] - mx);
      sum += e[t];
    }
#pragma unroll
    for (int m = 8; m > 0; m >>= 1) sum += __shfl_xor(sum, m);
    const float inv = 1.0f / sum;
#pragma unroll
    for (int t = 0; t < 8; ++t)
      P[row * ATT_STRIDE + sub * 8 + t] = (__bf16)(e[t] * inv);
  }
  __syncthreads();

  // O = P(16x128) * V(128x128); wave wid owns dh tile [wid*16, wid*16+16)
  v8f oac = (v8f){0.f, 0.f, 0.f, 0.f, 0.f, 0.f, 0.f, 0.f};
#pragma unroll
  for (int kc = 0; kc < 4; ++kc) {
    v16bf a  = load_frag(&P[kc * 32], lane, ATT_STRIDE);
    v16bf bb = load_frag(&Vt[(wid * 16) * ATT_STRIDE + kc * 32], lane, ATT_STRIDE);
    oac = wmma_bf16(a, bb, oac);
  }
#pragma unroll
  for (int r = 0; r < 8; ++r) {
    const int qrow = b * 16 + r + 8 * half;
    const int col  = h * 128 + wid * 16 + sub;
    Ob[(size_t)qrow * 2048 + col] = (__bf16)oac[r];
  }
}

// out = x (residual base; rows >= 256 stay exactly x)
__global__ __launch_bounds__(256) void copy_residual(
    const float* __restrict__ X, float* __restrict__ O) {
  const size_t i = (size_t)blockIdx.x * 256 + threadIdx.x;
  *(float4*)&O[i * 4] = *(const float4*)&X[i * 4];
}

// ---------------------------------------------------------------------------
extern "C" void kernel_launch(void* const* d_in, const int* in_sizes, int n_in,
                              void* d_out, int out_size, void* d_ws, size_t ws_size,
                              hipStream_t stream) {
  (void)in_sizes; (void)n_in; (void)out_size; (void)ws_size;
  const float* x  = (const float*)d_in[0];
  const float* Wq = (const float*)d_in[1];
  const float* Wk = (const float*)d_in[2];
  const float* Wv = (const float*)d_in[3];
  const float* Wo = (const float*)d_in[4];
  float* out = (float*)d_out;

  const int N = 2048, D = 2048, NQ = 256;

  char* wsp = (char*)d_ws;
  auto take = [&](size_t bytes) -> void* {
    void* p = (void*)wsp;
    wsp += (bytes + 255) & ~(size_t)255;
    return p;
  };
  __bf16* xnb = (__bf16*)take((size_t)N * D * 2);
  __bf16* WqT = (__bf16*)take((size_t)D * D * 2);
  __bf16* WkT = (__bf16*)take((size_t)D * D * 2);
  __bf16* WvT = (__bf16*)take((size_t)D * D * 2);
  __bf16* WoT = (__bf16*)take((size_t)D * D * 2);
  float*  Qf  = (float*)take((size_t)NQ * D * 4);
  float*  Kf  = (float*)take((size_t)N * D * 4);
  float*  Vf  = (float*)take((size_t)N * D * 4);
  __bf16* qb  = (__bf16*)take((size_t)NQ * D * 2);
  __bf16* kb  = (__bf16*)take((size_t)N * D * 2);
  __bf16* ob  = (__bf16*)take((size_t)NQ * D * 2);

  const dim3 tb(32, 8);
  transpose_to_bf16<<<dim3(64, 64), tb, 0, stream>>>(Wq, WqT, D);
  transpose_to_bf16<<<dim3(64, 64), tb, 0, stream>>>(Wk, WkT, D);
  transpose_to_bf16<<<dim3(64, 64), tb, 0, stream>>>(Wv, WvT, D);
  transpose_to_bf16<<<dim3(64, 64), tb, 0, stream>>>(Wo, WoT, D);

  ln_rows_bf16<<<N, 256, 0, stream>>>(x, xnb, D);

  // Q projection only needs the 256 query rows the mask permits.
  gemm_bf16_nt<<<dim3(16, 2),  256, 0, stream>>>(xnb, WqT, Qf, NQ, D, D, 0);
  gemm_bf16_nt<<<dim3(16, 16), 256, 0, stream>>>(xnb, WkT, Kf, N,  D, D, 0);
  gemm_bf16_nt<<<dim3(16, 16), 256, 0, stream>>>(xnb, WvT, Vf, N,  D, D, 0);

  head_ln_bf16<<<NQ * 16 / 8, 256, 0, stream>>>(Qf, qb, NQ);
  head_ln_bf16<<<N  * 16 / 8, 256, 0, stream>>>(Kf, kb, N);

  attention_kernel<<<dim3(16, 16), 256, 0, stream>>>(qb, kb, Vf, ob);

  copy_residual<<<(N * D / 4) / 256, 256, 0, stream>>>(x, out);
  // out rows 0..255 += O @ Wo  (rows >= 256 contribute zero by the mask)
  gemm_bf16_nt<<<dim3(16, 2), 256, 0, stream>>>(ob, WoT, out, NQ, D, D, 1);
}
